// fn_cls_33835752358231
// MI455X (gfx1250) — matrix operand order repaired and checked by reading, not compiled, not run on hardware
//
#include <hip/hip_runtime.h>
#include <hip/hip_bf16.h>
#include <math.h>

// ---------------- model constants ----------------
#define B_   2
#define S_   2048
#define D_   768
#define H_   12
#define DH_  64
#define L_   12
#define F_   3072
#define BS_  64
#define R_   3
#define NB_  32
#define NM_  30
#define M_   (B_ * S_)      // 4096 tokens
#define SCALE_ 0.125f       // 1/sqrt(64)

#define USE_TDM 1           // Tensor Data Mover for global->LDS tile staging

typedef __attribute__((ext_vector_type(16))) _Float16     v16h;
typedef __attribute__((ext_vector_type(8)))  float        v8f;
typedef __attribute__((ext_vector_type(4)))  unsigned int u32x4;
typedef __attribute__((ext_vector_type(8)))  int          i32x8;
typedef __attribute__((ext_vector_type(4)))  int          i32x4;

__device__ __forceinline__ v8f wmma16(v16h a, v16h b, v8f c) {
  return __builtin_amdgcn_wmma_f32_16x16x32_f16(false, a, false, b, (short)0, c,
                                                false, false);
}

// A operand (16x32, rows m0..m0+15, K kb..kb+31) from row-major f16 LDS tile.
// lane(m=l&15, hs=l>>4): elems 0..7 = halves kb+hs*8.., elems 8..15 = kb+16+hs*8..
__device__ __forceinline__ v16h frag_a16(const _Float16* base, int stride, int m0, int kb) {
  const int lane = threadIdx.x & 31;
  const int m = lane & 15, hs = lane >> 4;
  const _Float16* p = base + (m0 + m) * stride + kb + hs * 8;
  union { v16h h; u32x4 q[2]; } t;
  t.q[0] = *(const u32x4*)p;
  t.q[1] = *(const u32x4*)(p + 16);
  return t.h;
}

// B operand (32x16, cols n0..n0+15, K kb..kb+31) from B^T row-major [N][K] f16 LDS.
__device__ __forceinline__ v16h frag_b16(const _Float16* base, int stride, int n0, int kb) {
  const int lane = threadIdx.x & 31;
  const int n = lane & 15, hs = lane >> 4;
  const _Float16* p = base + (n0 + n) * stride + kb + hs * 8;
  union { v16h h; u32x4 q[2]; } t;
  t.q[0] = *(const u32x4*)p;
  t.q[1] = *(const u32x4*)(p + 16);
  return t.h;
}

__device__ __forceinline__ float gelu_f(float x) {
  float x3 = x * x * x;
  return 0.5f * x * (1.0f + tanhf(0.7978845608028654f * (x + 0.044715f * x3)));
}

#if USE_TDM
// 2D tile DMA: rows x rowlen f16 elements, row stride stride_elems, into LDS at
// lds_off; pad_amt+1 DWORDs of LDS padding appended per pad-interval (one row).
__device__ __forceinline__ void tdm_load_2d(unsigned lds_off, const void* gptr,
                                            unsigned rows, unsigned rowlen,
                                            unsigned stride_elems,
                                            unsigned pad_int, unsigned pad_amt) {
  unsigned long long ga = (unsigned long long)(uintptr_t)gptr;
  u32x4 g0;
  g0[0] = 1u;                                                   // count=1 (valid D#)
  g0[1] = lds_off;                                              // LDS byte address
  g0[2] = (unsigned)ga;                                         // global addr lo
  g0[3] = (unsigned)((ga >> 32) & 0x01FFFFFFull) | (2u << 30);  // addr[56:32] | type=2
  i32x8 g1;
  g1[0] = (int)((1u << 16) | (1u << 20) | (pad_int << 22) | (pad_amt << 25));
  //            data_size=2B  pad_enable    pad_interval        pad_amount
  g1[1] = (int)((stride_elems & 0xFFFFu) << 16);                 // tensor_dim0 lo16
  g1[2] = (int)((stride_elems >> 16) | ((rows & 0xFFFFu) << 16));// dim0 hi | tensor_dim1 lo
  g1[3] = (int)((rows >> 16) | (rowlen << 16));                  // dim1 hi | tile_dim0
  g1[4] = (int)rows;                                             // tile_dim1 (tile_dim2=0)
  g1[5] = (int)stride_elems;                                     // tensor_dim0_stride lo32
  g1[6] = 0;
  g1[7] = 0;
  i32x4 z4 = {0, 0, 0, 0};
#if defined(__clang_major__) && __clang_major__ >= 23
  i32x8 z8 = {0, 0, 0, 0, 0, 0, 0, 0};
  __builtin_amdgcn_tensor_load_to_lds(g0, g1, z4, z4, z8, 0);
#else
  __builtin_amdgcn_tensor_load_to_lds(g0, g1, z4, z4, 0);
#endif
}
template <int N>
__device__ __forceinline__ void tdm_wait() { __builtin_amdgcn_s_wait_tensorcnt(N); }
__device__ __forceinline__ unsigned lds_addr_of(const void* p) {
  return (unsigned)(uintptr_t)p;   // generic LDS addr: low 32 bits = LDS offset
}
#endif

// ---------------- weight transpose+convert: in[K][N] f32 -> out[N][K] f16 ----------------
__global__ __launch_bounds__(256) void convT_kernel(const float* __restrict__ in,
                                                    _Float16* __restrict__ out,
                                                    int K, int N) {
  __shared__ float t[32][33];
  const int bx = blockIdx.x, by = blockIdx.y;        // bx: N/32, by: K/32
  const int tx = threadIdx.x & 31, ty = threadIdx.x >> 5;
#pragma unroll
  for (int j = 0; j < 4; ++j) {
    int r = ty + j * 8;
    t[r][tx] = in[(size_t)(by * 32 + r) * N + bx * 32 + tx];
  }
  __syncthreads();
#pragma unroll
  for (int j = 0; j < 4; ++j) {
    int r = ty + j * 8;
    out[(size_t)(bx * 32 + r) * K + by * 32 + tx] = (_Float16)t[tx][r];
  }
}

// ---------------- GEMM: out[M,N] = act(A[M,K](f16) * Wt[N,K](f16)^T + bias) ----------------
// OMODE: 0 = f32 row-major, 1 = f16 row-major, 2 = f16 row-major + GELU,
//        3 = f16 scattered to [B][H][DH][S] (V projection).
// grid = (N/128, M/64), block = 128: 4 waves, each a 32x64 tile (2x4 WMMA tiles).
// TDM double-buffered software pipeline: issue tiles(s+1), s_wait_tensorcnt 2, compute(s).
template <int OMODE>
__global__ __launch_bounds__(128) void gemm16_kernel(
    const _Float16* __restrict__ A, const _Float16* __restrict__ W,
    const float* __restrict__ bias, void* __restrict__ out, int Nn, int Kk) {
  __shared__ __align__(16) _Float16 As[2][64 * 40];    // [m][k], stride 40 halves
  __shared__ __align__(16) _Float16 Ws[2][128 * 40];   // [n][k]
  const int nT = blockIdx.x, mT = blockIdx.y;
  const int tid = threadIdx.x;
  const int wid = tid >> 5, lane = tid & 31;
  const int wr = wid >> 1, wc = wid & 1;
  const int nsteps = Kk >> 5;

  v8f acc[2][4];
#pragma unroll
  for (int i = 0; i < 2; ++i)
#pragma unroll
    for (int j = 0; j < 4; ++j) acc[i][j] = (v8f){0, 0, 0, 0, 0, 0, 0, 0};

#if USE_TDM
  if (wid == 0) {  // prologue: stage step 0
    tdm_load_2d(lds_addr_of(As[0]), A + (size_t)(mT * 64) * Kk, 64, 32, (unsigned)Kk, 3u, 3u);
    tdm_load_2d(lds_addr_of(Ws[0]), W + (size_t)(nT * 128) * Kk, 128, 32, (unsigned)Kk, 3u, 3u);
  }
#endif

  for (int s = 0; s < nsteps; ++s) {
    const int cur = s & 1;
#if USE_TDM
    if (wid == 0) {
      if (s + 1 < nsteps) {  // prefetch next tiles into the other buffer
        const int kk = (s + 1) << 5;
        tdm_load_2d(lds_addr_of(As[cur ^ 1]), A + (size_t)(mT * 64) * Kk + kk, 64, 32,
                    (unsigned)Kk, 3u, 3u);
        tdm_load_2d(lds_addr_of(Ws[cur ^ 1]), W + (size_t)(nT * 128) * Kk + kk, 128, 32,
                    (unsigned)Kk, 3u, 3u);
        tdm_wait<2>();   // step-s tiles complete; step-(s+1) pair may stay in flight
      } else {
        tdm_wait<0>();
      }
    }
#else
    const int kk = s << 5;
#pragma unroll
    for (int i = 0; i < 2; ++i) {
      int idx = tid + i * 128;
      int r = idx >> 2, c = idx & 3;
      *(u32x4*)(As[cur] + r * 40 + c * 8) =
          *(const u32x4*)(A + (size_t)(mT * 64 + r) * Kk + kk + c * 8);
    }
#pragma unroll
    for (int i = 0; i < 4; ++i) {
      int idx = tid + i * 128;
      int r = idx >> 2, c = idx & 3;
      *(u32x4*)(Ws[cur] + r * 40 + c * 8) =
          *(const u32x4*)(W + (size_t)(nT * 128 + r) * Kk + kk + c * 8);
    }
#endif
    __syncthreads();   // (A) tiles of step s visible to all waves
    v16h af[2], bf[4];
#pragma unroll
    for (int i = 0; i < 2; ++i) af[i] = frag_a16(As[cur], 40, wr * 32 + i * 16, 0);
#pragma unroll
    for (int j = 0; j < 4; ++j) bf[j] = frag_b16(Ws[cur], 40, wc * 64 + j * 16, 0);
#pragma unroll
    for (int i = 0; i < 2; ++i)
#pragma unroll
      for (int j = 0; j < 4; ++j) acc[i][j] = wmma16(af[i], bf[j], acc[i][j]);
    __syncthreads();   // (B) buffer free for reuse
  }

  const int n = lane & 15, hs = lane >> 4;
#pragma unroll
  for (int i = 0; i < 2; ++i)
#pragma unroll
    for (int j = 0; j < 4; ++j) {
      int col = nT * 128 + wc * 64 + j * 16 + n;
      float bv = bias[col];
#pragma unroll
      for (int r = 0; r < 8; ++r) {
        int row = mT * 64 + wr * 32 + i * 16 + hs * 8 + r;
        float vv = acc[i][j][r] + bv;
        if (OMODE == 0) {
          ((float*)out)[(size_t)row * Nn + col] = vv;
        } else if (OMODE == 1) {
          ((_Float16*)out)[(size_t)row * Nn + col] = (_Float16)vv;
        } else if (OMODE == 2) {
          ((_Float16*)out)[(size_t)row * Nn + col] = (_Float16)gelu_f(vv);
        } else {  // V projection -> [B][H][DH][S]
          int bb = row >> 11, ss = row & (S_ - 1);
          int hh = col >> 6, dh = col & 63;
          ((_Float16*)out)[(((size_t)bb * H_ + hh) * DH_ + dh) * S_ + ss] = (_Float16)vv;
        }
      }
    }
}

// ---------------- fused BigBird block attention (flash-style online softmax) ----------------
// grid = (NB, H, B), block = 128. q,k: [B*S][D] f16; vT: [B][H][DH][S] f16.
// K/V tiles double-buffered through the Tensor Data Mover.
__global__ __launch_bounds__(128) void bigbird_attn_kernel(
    const _Float16* __restrict__ q, const _Float16* __restrict__ k,
    const _Float16* __restrict__ vT, const int* __restrict__ mask,
    const int* __restrict__ rand_blocks, _Float16* __restrict__ ctx) {
  __shared__ __align__(16) _Float16 Qs[64 * 72];        // [query][dh]
  __shared__ __align__(16) _Float16 Ks[2][64 * 72];     // [key][dh]
  __shared__ __align__(16) _Float16 Vt[2][64 * 72];     // [dh][key]
  __shared__ __align__(16) _Float16 Ps[4 * 16 * 72];    // per-wave P [q][key]
  __shared__ float negs[2][64];
  __shared__ int klist[32];

  const int qb = blockIdx.x, head = blockIdx.y, b = blockIdx.z;
  const int tid = threadIdx.x, wid = tid >> 5, lane = tid & 31;
  const int nq = lane & 15, hs = lane >> 4;

  int KL;
  if (qb == 0 || qb == NB_ - 1) {        // global query blocks attend everywhere
    KL = NB_;
    if (tid < NB_) klist[tid] = tid;
  } else {                               // window + global + random key blocks
    KL = 8;
    if (tid == 0) {
      klist[0] = qb - 1; klist[1] = qb; klist[2] = qb + 1;
      klist[3] = 0;      klist[4] = NB_ - 1;
      const int* rb = rand_blocks + ((size_t)head * NM_ + (qb - 1)) * R_;
      klist[5] = rb[0]; klist[6] = rb[1]; klist[7] = rb[2];
    }
  }

#if USE_TDM
  if (wid == 0) {   // klist written entirely by wave 0 -> in-wave DS ordering suffices
    tdm_load_2d(lds_addr_of(Qs), q + (size_t)(b * S_ + qb * 64) * D_ + head * DH_,
                64, 64, (unsigned)D_, 4u, 3u);
    const int k0 = klist[0];
    tdm_load_2d(lds_addr_of(Ks[0]), k + (size_t)(b * S_ + k0 * 64) * D_ + head * DH_,
                64, 64, (unsigned)D_, 4u, 3u);
    tdm_load_2d(lds_addr_of(Vt[0]), vT + ((size_t)(b * H_ + head) * DH_) * S_ + k0 * 64,
                64, 64, (unsigned)S_, 4u, 3u);
  }
  __syncthreads();                       // klist visible to all waves
  if (tid < 64) negs[0][tid] = (1 - mask[b * S_ + klist[0] * 64 + tid]) * -1e9f;
#else
#pragma unroll
  for (int i = 0; i < 4; ++i) {
    int idx = tid + i * 128;
    int row = idx >> 3, c = idx & 7;
    *(u32x4*)(Qs + row * 72 + c * 8) =
        *(const u32x4*)(q + (size_t)(b * S_ + qb * 64 + row) * D_ + head * DH_ + c * 8);
  }
  __syncthreads();                       // klist + Qs visible
#endif

  _Float16* Pw = Ps + wid * (16 * 72);
  float m_run = -1e30f, l_run = 0.0f;
  v8f o[4];
#pragma unroll
  for (int t = 0; t < 4; ++t) o[t] = (v8f){0, 0, 0, 0, 0, 0, 0, 0};

  for (int s = 0; s < KL; ++s) {
    const int cur = s & 1;
#if USE_TDM
    if (wid == 0) {
      if (s + 1 < KL) {                  // prefetch next K/V pair
        const int kn = klist[s + 1];
        tdm_load_2d(lds_addr_of(Ks[cur ^ 1]),
                    k + (size_t)(b * S_ + kn * 64) * D_ + head * DH_,
                    64, 64, (unsigned)D_, 4u, 3u);
        tdm_load_2d(lds_addr_of(Vt[cur ^ 1]),
                    vT + ((size_t)(b * H_ + head) * DH_) * S_ + kn * 64,
                    64, 64, (unsigned)S_, 4u, 3u);
        tdm_wait<2>();                   // step-s tiles (and Qs) complete
      } else {
        tdm_wait<0>();
      }
    }
    if (tid < 64 && s + 1 < KL)
      negs[cur ^ 1][tid] = (1 - mask[b * S_ + klist[s + 1] * 64 + tid]) * -1e9f;
#else
    const int kblk = klist[s];
#pragma unroll
    for (int i = 0; i < 4; ++i) {
      int idx = tid + i * 128;
      int row = idx >> 3, c = idx & 7;
      *(u32x4*)(Ks[cur] + row * 72 + c * 8) =
          *(const u32x4*)(k + (size_t)(b * S_ + kblk * 64 + row) * D_ + head * DH_ + c * 8);
      *(u32x4*)(Vt[cur] + row * 72 + c * 8) =
          *(const u32x4*)(vT + ((size_t)(b * H_ + head) * DH_ + row) * S_ + kblk * 64 + c * 8);
    }
    if (tid < 64) negs[cur][tid] = (1 - mask[b * S_ + kblk * 64 + tid]) * -1e9f;
#endif
    __syncthreads();                     // (A) step-s tiles visible

    // S^T = K * Q^T : lane column = query nq, rows = keys kt*16 + hs*8 + r
    float sc[4][8];
#pragma unroll
    for (int kt = 0; kt < 4; ++kt) {
      v8f c = (v8f){0, 0, 0, 0, 0, 0, 0, 0};
      c = wmma16(frag_a16(Ks[cur], 72, kt * 16, 0),  frag_b16(Qs, 72, wid * 16, 0),  c);
      c = wmma16(frag_a16(Ks[cur], 72, kt * 16, 32), frag_b16(Qs, 72, wid * 16, 32), c);
#pragma unroll
      for (int r = 0; r < 8; ++r)
        sc[kt][r] = c[r] * SCALE_ + negs[cur][kt * 16 + hs * 8 + r];
    }

    float tm = -1e30f;
#pragma unroll
    for (int kt = 0; kt < 4; ++kt)
#pragma unroll
      for (int r = 0; r < 8; ++r) tm = fmaxf(tm, sc[kt][r]);
    tm = fmaxf(tm, __shfl_xor(tm, 16));  // lane pair shares one query column
    float mnew = fmaxf(m_run, tm);
    float fac = __expf(m_run - mnew);
    m_run = mnew;
    l_run *= fac;
#pragma unroll
    for (int kt = 0; kt < 4; ++kt) {
      union { u32x4 qv; _Float16 hh[8]; } pk;
#pragma unroll
      for (int r = 0; r < 8; ++r) {
        float p = __expf(sc[kt][r] - mnew);
        l_run += p;
        pk.hh[r] = (_Float16)p;
      }
      *(u32x4*)(Pw + nq * 72 + kt * 16 + hs * 8) = pk.qv;  // one b128 store
    }
#pragma unroll
    for (int r = 0; r < 8; ++r) {        // rescale O (rows are queries hs*8+r)
      float fq = __shfl(fac, hs * 8 + r);
#pragma unroll
      for (int t = 0; t < 4; ++t) o[t][r] *= fq;
    }
#pragma unroll
    for (int kp = 0; kp < 2; ++kp) {     // O += P[16q x 64keys] * V[64keys x 64dh]
      v16h pa = frag_a16(Pw, 72, 0, kp * 32);
#pragma unroll
      for (int t = 0; t < 4; ++t)
        o[t] = wmma16(pa, frag_b16(Vt[cur], 72, t * 16, kp * 32), o[t]);
    }
    __syncthreads();                     // (B) buffer free for prefetch reuse
  }

  float ltot = l_run + __shfl_xor(l_run, 16);
#pragma unroll
  for (int r = 0; r < 8; ++r) {
    float linv = 1.0f / __shfl(ltot, hs * 8 + r);
    int srow = qb * 64 + wid * 16 + hs * 8 + r;
    size_t base = (size_t)(b * S_ + srow) * D_ + head * DH_;
#pragma unroll
    for (int t = 0; t < 4; ++t)
      ctx[base + t * 16 + nq] = (_Float16)(o[t][r] * linv);
  }
}

// ---------------- embedding gather + LayerNorm (writes f32 h and f16 shadow) ----------------
__global__ __launch_bounds__(256) void embed_ln_kernel(
    const int* __restrict__ ids, const float* __restrict__ ew,
    const float* __restrict__ ep, const float* __restrict__ g,
    const float* __restrict__ bta, float* __restrict__ out,
    _Float16* __restrict__ out16) {
  __shared__ float r1[256], r2[256];
  const int tok = blockIdx.x;
  const int pos = tok & (S_ - 1);
  const int id = ids[tok];
  const int t = threadIdx.x;
  float vals[3], s1 = 0.f, s2 = 0.f;
#pragma unroll
  for (int i = 0; i < 3; ++i) {
    int d = t + i * 256;
    float vv = ew[(size_t)id * D_ + d] + ep[(size_t)pos * D_ + d];
    vals[i] = vv; s1 += vv; s2 += vv * vv;
  }
  r1[t] = s1; r2[t] = s2; __syncthreads();
  for (int off = 128; off > 0; off >>= 1) {
    if (t < off) { r1[t] += r1[t + off]; r2[t] += r2[t + off]; }
    __syncthreads();
  }
  float mean = r1[0] * (1.0f / D_);
  float var  = r2[0] * (1.0f / D_) - mean * mean;
  float inv  = rsqrtf(var + 1e-12f);
#pragma unroll
  for (int i = 0; i < 3; ++i) {
    int d = t + i * 256;
    float o = (vals[i] - mean) * inv * g[d] + bta[d];
    out[(size_t)tok * D_ + d] = o;
    out16[(size_t)tok * D_ + d] = (_Float16)o;
  }
}

// ---------------- residual add + LayerNorm (in-place safe; writes f32 + f16) ----------------
__global__ __launch_bounds__(256) void add_ln_kernel(
    const float* __restrict__ xin, const float* __restrict__ resid,
    const float* __restrict__ g, const float* __restrict__ bta,
    float* __restrict__ out, _Float16* __restrict__ out16) {
  __shared__ float r1[256], r2[256];
  const int tok = blockIdx.x;
  const int t = threadIdx.x;
  float vals[3], s1 = 0.f, s2 = 0.f;
#pragma unroll
  for (int i = 0; i < 3; ++i) {
    int d = t + i * 256;
    float vv = xin[(size_t)tok * D_ + d] + resid[(size_t)tok * D_ + d];
    vals[i] = vv; s1 += vv; s2 += vv * vv;
  }
  r1[t] = s1; r2[t] = s2; __syncthreads();
  for (int off = 128; off > 0; off >>= 1) {
    if (t < off) { r1[t] += r1[t + off]; r2[t] += r2[t + off]; }
    __syncthreads();
  }
  float mean = r1[0] * (1.0f / D_);
  float var  = r2[0] * (1.0f / D_) - mean * mean;
  float inv  = rsqrtf(var + 1e-12f);
#pragma unroll
  for (int i = 0; i < 3; ++i) {
    int d = t + i * 256;
    float o = (vals[i] - mean) * inv * g[d] + bta[d];
    out[(size_t)tok * D_ + d] = o;
    out16[(size_t)tok * D_ + d] = (_Float16)o;
  }
}

// ---------------- pooler ----------------
__global__ __launch_bounds__(256) void pool_kernel(
    const float* __restrict__ h, const float* __restrict__ pw,
    const float* __restrict__ pb, float* __restrict__ pooled) {
  int idx = blockIdx.x * blockDim.x + threadIdx.x;
  if (idx >= 2 * D_) return;
  int b = idx / D_, j = idx - b * D_;
  const float* row = h + (size_t)(b * S_) * D_;
  float acc = pb[j];
  for (int d = 0; d < D_; ++d) acc += row[d] * pw[(size_t)d * D_ + j];
  pooled[idx] = tanhf(acc);
}

// ---------------- classifier ----------------
__global__ void cls_kernel(const float* __restrict__ pooled,
                           const float* __restrict__ cw,
                           const float* __restrict__ cb,
                           float* __restrict__ out) {
  int i = threadIdx.x;
  if (i < 4) {
    int b = i >> 1, c = i & 1;
    float acc = cb[c];
    for (int d = 0; d < D_; ++d) acc += pooled[(size_t)b * D_ + d] * cw[(size_t)d * 2 + c];
    out[i] = acc;
  }
}

// ---------------- host orchestration ----------------
extern "C" void kernel_launch(void* const* d_in, const int* in_sizes, int n_in,
                              void* d_out, int out_size, void* d_ws, size_t ws_size,
                              hipStream_t stream) {
  (void)in_sizes; (void)n_in; (void)out_size; (void)ws_size;

  const int*   x_ids  = (const int*)d_in[0];
  const int*   amask  = (const int*)d_in[1];
  const int*   rand_b = (const int*)d_in[2];
  const float* emb_w  = (const float*)d_in[3];
  const float* emb_p  = (const float*)d_in[4];
  const float* eln_g  = (const float*)d_in[5];
  const float* eln_b  = (const float*)d_in[6];
  const float* wq = (const float*)d_in[7];
  const float* bq = (const float*)d_in[8];
  const float* wk = (const float*)d_in[9];
  const float* bk = (const float*)d_in[10];
  const float* wv = (const float*)d_in[11];
  const float* bv = (const float*)d_in[12];
  const float* wo = (const float*)d_in[13];
  const float* bo = (const float*)d_in[14];
  const float* ln1g = (const float*)d_in[15];
  const float* ln1b = (const float*)d_in[16];
  const float* w1 = (const float*)d_in[17];
  const float* b1 = (const float*)d_in[18];
  const float* w2 = (const float*)d_in[19];
  const float* b2 = (const float*)d_in[20];
  const float* ln2g = (const float*)d_in[21];
  const float* ln2b = (const float*)d_in[22];
  const float* poolw = (const float*)d_in[23];
  const float* poolb = (const float*)d_in[24];
  const float* clsw  = (const float*)d_in[25];
  const float* clsb  = (const float*)d_in[26];

  const size_t MD = (size_t)M_ * D_;
  const size_t DD = (size_t)D_ * D_;
  const size_t DF = (size_t)D_ * F_;

  // workspace: h(f32) | buf2(f32) | h16 | act16 (q|k|vT|ctx, later ffn1) | weight staging
  float*    h     = (float*)d_ws;
  float*    buf2  = h + MD;
  _Float16* h16   = (_Float16*)(buf2 + MD);
  _Float16* act16 = h16 + MD;
  _Float16* wst   = act16 + (size_t)M_ * F_;
  _Float16* q16   = act16;
  _Float16* k16   = act16 + MD;
  _Float16* vT16  = act16 + 2 * MD;
  _Float16* c16   = act16 + 3 * MD;
  _Float16* ffn16 = act16;               // reuses q|k|vT|ctx region
  _Float16* wq16 = wst;
  _Float16* wk16 = wst + DD;
  _Float16* wv16 = wst + 2 * DD;
  _Float16* wo16 = wst + 3 * DD;
  _Float16* w116 = wst + 4 * DD;         // [F][D]
  _Float16* w216 = w116 + DF;            // [D][F]
  float* pooled = buf2;                  // free after final add_ln

  const dim3 gP(D_ / 128, M_ / 64);      // 6 x 64
  const dim3 gF(F_ / 128, M_ / 64);      // 24 x 64
  const dim3 gA(NB_, H_, B_);
  const dim3 tSq(D_ / 32, D_ / 32);      // 24 x 24
  const dim3 t1(F_ / 32, D_ / 32);       // 96 x 24  (w1: [D][F] -> [F][D])
  const dim3 t2(D_ / 32, F_ / 32);       // 24 x 96  (w2: [F][D] -> [D][F])

  embed_ln_kernel<<<M_, 256, 0, stream>>>(x_ids, emb_w, emb_p, eln_g, eln_b, h, h16);

  for (int l = 0; l < L_; ++l) {
    const size_t od = (size_t)l * DD;
    convT_kernel<<<tSq, 256, 0, stream>>>(wq + od, wq16, D_, D_);
    convT_kernel<<<tSq, 256, 0, stream>>>(wk + od, wk16, D_, D_);
    convT_kernel<<<tSq, 256, 0, stream>>>(wv + od, wv16, D_, D_);
    convT_kernel<<<tSq, 256, 0, stream>>>(wo + od, wo16, D_, D_);
    convT_kernel<<<t1, 256, 0, stream>>>(w1 + (size_t)l * DF, w116, D_, F_);
    convT_kernel<<<t2, 256, 0, stream>>>(w2 + (size_t)l * DF, w216, F_, D_);

    gemm16_kernel<1><<<gP, 128, 0, stream>>>(h16, wq16, bq + (size_t)l * D_, q16, D_, D_);
    gemm16_kernel<1><<<gP, 128, 0, stream>>>(h16, wk16, bk + (size_t)l * D_, k16, D_, D_);
    gemm16_kernel<3><<<gP, 128, 0, stream>>>(h16, wv16, bv + (size_t)l * D_, vT16, D_, D_);
    bigbird_attn_kernel<<<gA, 128, 0, stream>>>(q16, k16, vT16, amask, rand_b, c16);
    gemm16_kernel<0><<<gP, 128, 0, stream>>>(c16, wo16, bo + (size_t)l * D_, buf2, D_, D_);
    add_ln_kernel<<<M_, 256, 0, stream>>>(h, buf2, ln1g + (size_t)l * D_, ln1b + (size_t)l * D_, h, h16);
    gemm16_kernel<2><<<gF, 128, 0, stream>>>(h16, w116, b1 + (size_t)l * F_, ffn16, F_, D_);
    gemm16_kernel<0><<<gP, 128, 0, stream>>>(ffn16, w216, b2 + (size_t)l * D_, buf2, D_, F_);
    add_ln_kernel<<<M_, 256, 0, stream>>>(h, buf2, ln2g + (size_t)l * D_, ln2b + (size_t)l * D_, h, h16);
  }

  pool_kernel<<<(2 * D_ + 255) / 256, 256, 0, stream>>>(h, poolw, poolb, pooled);
  cls_kernel<<<1, 32, 0, stream>>>(pooled, clsw, clsb, (float*)d_out);
}
